// PatchCore_16896401342573
// MI455X (gfx1250) — matrix operand order repaired and checked by reading, not compiled, not run on hardware
//
#include <hip/hip_runtime.h>
#include <math.h>

// ---------------------------------------------------------------------------
// PatchCore anomaly scoring for MI455X (gfx1250, wave32, WMMA).
// Heavy phase: 784 x 100000 x 1536 distance GEMM via v_wmma_f32_16x16x32_bf16.
// 2 M-tiles x 4 N-tiles per wave; K-loop unrolled x2 with ping-pong fragment
// register sets (true double buffering: no rotation movs, no WAR-hazard NOPs).
// ---------------------------------------------------------------------------

typedef __attribute__((ext_vector_type(16))) __bf16 v16bf;
typedef __attribute__((ext_vector_type(8)))  __bf16 v8bf;
typedef __attribute__((ext_vector_type(8)))  float   v8f;
typedef __attribute__((ext_vector_type(4)))  float   v4f;

#define PP        784
#define DD        1536
#define NLIB      100000
#define NCHUNK    64
#define NBLK      1563              // ceil(100000 / 64)
#define LDS_ROW   1544              // 1536 + 8 bf16 pad -> bank spread
#define IMGW      224
#define IMGN      (IMGW * IMGW)

__device__ __forceinline__ v16bf cat16(v8bf lo, v8bf hi) {
  v16bf r;
#pragma unroll
  for (int i = 0; i < 8; ++i) { r[i] = lo[i]; r[i + 8] = hi[i]; }
  return r;
}

// Load one K-step's fragments (2 A tiles from global/L2, 4 B tiles from LDS).
__device__ __forceinline__ void load_frags(
    const __bf16* __restrict__ A0, const __bf16* __restrict__ A1,
    const __bf16* __restrict__ Bbase, int k, int half,
    v8bf& a0l, v8bf& a0h, v8bf& a1l, v8bf& a1h,
    v8bf* __restrict__ bl, v8bf* __restrict__ bh) {
  a0l = *(const v8bf*)(A0 + k + half * 8);
  a0h = *(const v8bf*)(A0 + k + 16 + half * 8);
  a1l = *(const v8bf*)(A1 + k + half * 8);
  a1h = *(const v8bf*)(A1 + k + 16 + half * 8);
#pragma unroll
  for (int nt = 0; nt < 4; ++nt) {
    const __bf16* br = Bbase + nt * 16 * LDS_ROW + k;
    bl[nt] = *(const v8bf*)(br + half * 8);
    bh[nt] = *(const v8bf*)(br + 16 + half * 8);
  }
}

// 8 WMMAs for one K-step: 2 M-tiles x 4 N-tiles, each B frag feeds 2 WMMAs.
__device__ __forceinline__ void do_wmma(
    v8f acc[2][4],
    v8bf a0l, v8bf a0h, v8bf a1l, v8bf a1h,
    const v8bf* __restrict__ bl, const v8bf* __restrict__ bh) {
  const v16bf fa0 = cat16(a0l, a0h);
  const v16bf fa1 = cat16(a1l, a1h);
#pragma unroll
  for (int nt = 0; nt < 4; ++nt) {
    const v16bf fb = cat16(bl[nt], bh[nt]);
    acc[0][nt] = __builtin_amdgcn_wmma_f32_16x16x32_bf16(
        false, fa0, false, fb, (short)0, acc[0][nt], false, false);
    acc[1][nt] = __builtin_amdgcn_wmma_f32_16x16x32_bf16(
        false, fa1, false, fb, (short)0, acc[1][nt], false, false);
  }
}

// ---------------------------------------------------------------------------
// 1) patch f32 -> bf16 (staged in ws) + row norms ||x||^2
// ---------------------------------------------------------------------------
__global__ void k_convert_patch(const float* __restrict__ patch,
                                __bf16* __restrict__ patch_bf,
                                float* __restrict__ x2) {
  __shared__ float red[256];
  const int row = blockIdx.x;
  const float* src = patch + (size_t)row * DD;
  __bf16* dst = patch_bf + (size_t)row * DD;
  float acc = 0.f;
  for (int j = threadIdx.x; j < DD; j += 256) {
    float v = src[j];
    acc += v * v;
    dst[j] = (__bf16)v;
  }
  red[threadIdx.x] = acc;
  __syncthreads();
  for (int s = 128; s > 0; s >>= 1) {
    if (threadIdx.x < s) red[threadIdx.x] += red[threadIdx.x + s];
    __syncthreads();
  }
  if (threadIdx.x == 0) x2[row] = red[0];
}

// ---------------------------------------------------------------------------
// 2) Fused distance GEMM + per-row min/argmin partials.
//    Block: 64 library rows staged once into LDS as bf16 (193KB of 320KB).
// ---------------------------------------------------------------------------
__global__ void __launch_bounds__(256, 1)
k_dist_min(const float* __restrict__ lib,
           const __bf16* __restrict__ patch_bf,
           const float* __restrict__ x2,
           float* __restrict__ pval,
           int* __restrict__ pidx) {
  extern __shared__ char smem[];
  float*  s_y2  = (float*)smem;               // 64 norms
  __bf16* s_lib = (__bf16*)(smem + 256);      // 64 * LDS_ROW bf16

  const int tid = threadIdx.x;
  const int n0  = blockIdx.x * NCHUNK;

  if (tid < NCHUNK) s_y2[tid] = 0.f;
  __syncthreads();

  // ---- stage: 64 lib rows f32 -> bf16, accumulate ||y||^2 ----
  {
    const int r   = tid >> 2;   // 0..63
    const int seg = tid & 3;    // quarter of the row
    const int n   = n0 + r;
    __bf16* drow  = s_lib + r * LDS_ROW + seg * 384;
    float part = 0.f;
    if (n < NLIB) {
      const float* srow = lib + (size_t)n * DD + seg * 384;
      for (int j = 0; j < 384; j += 8) {
        v4f a = *(const v4f*)(srow + j);
        v4f b = *(const v4f*)(srow + j + 4);
        part += a[0]*a[0] + a[1]*a[1] + a[2]*a[2] + a[3]*a[3]
              + b[0]*b[0] + b[1]*b[1] + b[2]*b[2] + b[3]*b[3];
        v8bf o;
        o[0]=(__bf16)a[0]; o[1]=(__bf16)a[1]; o[2]=(__bf16)a[2]; o[3]=(__bf16)a[3];
        o[4]=(__bf16)b[0]; o[5]=(__bf16)b[1]; o[6]=(__bf16)b[2]; o[7]=(__bf16)b[3];
        *(v8bf*)(drow + j) = o;
      }
    } else {
      v8bf z;
      #pragma unroll
      for (int i = 0; i < 8; ++i) z[i] = (__bf16)0.0f;
      for (int j = 0; j < 384; j += 8) *(v8bf*)(drow + j) = z;
      if (seg == 0) part = 1e30f;            // push OOB rows out of the min
    }
    atomicAdd(&s_y2[r], part);
  }
  __syncthreads();

  const int wave   = tid >> 5;
  const int lane   = tid & 31;
  const int lanelo = lane & 15;
  const int half   = lane >> 4;

  for (int iter = 0; iter < 4; ++iter) {
    const int mt0 = (iter * 8 + wave) * 2;   // wave-uniform, even
    if (mt0 >= 49) break;
    const int mt1 = mt0 + 1;                 // may be 49 (discarded)

    v8f acc[2][4];
    #pragma unroll
    for (int u = 0; u < 2; ++u)
      #pragma unroll
      for (int nt = 0; nt < 4; ++nt) { v8f z = {}; acc[u][nt] = z; }

    // A layout (16-bit 16x32): lanes 0-15 hold K {0..7,16..23}, lanes 16-31
    // hold K {8..15,24..31}, M = lane&15.
    const __bf16* A0 = patch_bf + (size_t)(mt0 * 16 + lanelo) * DD;
    const __bf16* A1 = patch_bf +
        (size_t)((mt1 < 49 ? mt1 : mt0) * 16 + lanelo) * DD;
    // B layout: column = lane&15, K {0..7,16..23} (lanes 0-15) / {8..15,24..31} (lanes 16-31)
    const __bf16* Bbase = s_lib + lanelo * LDS_ROW;

    // ---- ping-pong fragment register sets ----
    v8bf a0l0, a0h0, a1l0, a1h0, bl0[4], bh0[4];   // set 0
    v8bf a0l1, a0h1, a1l1, a1h1, bl1[4], bh1[4];   // set 1

    load_frags(A0, A1, Bbase, 0, half, a0l0, a0h0, a1l0, a1h0, bl0, bh0);

    // steady state: 23 iterations of 2 K-steps, no copies between sets
    for (int k0 = 0; k0 < DD - 64; k0 += 64) {
      load_frags(A0, A1, Bbase, k0 + 32, half, a0l1, a0h1, a1l1, a1h1, bl1, bh1);
      do_wmma(acc, a0l0, a0h0, a1l0, a1h0, bl0, bh0);
      load_frags(A0, A1, Bbase, k0 + 64, half, a0l0, a0h0, a1l0, a1h0, bl0, bh0);
      do_wmma(acc, a0l1, a0h1, a1l1, a1h1, bl1, bh1);
    }
    // tail: k0 = DD-64 (fragments for DD-64 already in set 0)
    load_frags(A0, A1, Bbase, DD - 32, half, a0l1, a0h1, a1l1, a1h1, bl1, bh1);
    do_wmma(acc, a0l0, a0h0, a1l0, a1h0, bl0, bh0);
    do_wmma(acc, a0l1, a0h1, a1l1, a1h1, bl1, bh1);

    // Epilogue: d = sqrt(max(x2+y2-2*dot,0)); min over this block's 64 cols.
    const float y2v0 = s_y2[ 0 + lanelo];
    const float y2v1 = s_y2[16 + lanelo];
    const float y2v2 = s_y2[32 + lanelo];
    const float y2v3 = s_y2[48 + lanelo];

    #pragma unroll
    for (int u = 0; u < 2; ++u) {
      const int mt = mt0 + u;
      if (mt >= 49) continue;                  // wave-uniform
      #pragma unroll
      for (int r = 0; r < 8; ++r) {
        const int   p  = mt * 16 + half * 8 + r;  // C layout: M = r + 8*half
        const float xp = x2[p];

        float d  = sqrtf(fmaxf(xp + y2v0 - 2.0f * acc[u][0][r], 0.f));
        float best = d; int bi = n0 + lanelo;
        d = sqrtf(fmaxf(xp + y2v1 - 2.0f * acc[u][1][r], 0.f));
        if (d < best) { best = d; bi = n0 + 16 + lanelo; }
        d = sqrtf(fmaxf(xp + y2v2 - 2.0f * acc[u][2][r], 0.f));
        if (d < best) { best = d; bi = n0 + 32 + lanelo; }
        d = sqrtf(fmaxf(xp + y2v3 - 2.0f * acc[u][3][r], 0.f));
        if (d < best) { best = d; bi = n0 + 48 + lanelo; }

        #pragma unroll
        for (int m = 1; m < 16; m <<= 1) {     // reduce within each 16-lane half
          float ov = __shfl_xor(best, m, 32);
          int   oi = __shfl_xor(bi,   m, 32);
          if (ov < best || (ov == best && oi < bi)) { best = ov; bi = oi; }
        }
        if (lanelo == 0) {
          pval[(size_t)p * NBLK + blockIdx.x] = best;
          pidx[(size_t)p * NBLK + blockIdx.x] = bi;
        }
      }
    }
  }
}

// ---------------------------------------------------------------------------
// 3) final min/argmin across the 1563 block partials (one block per row p)
// ---------------------------------------------------------------------------
__global__ void k_reduce_min(const float* __restrict__ pval,
                             const int* __restrict__ pidx,
                             float* __restrict__ minv, int* __restrict__ mini) {
  __shared__ float sv[256];
  __shared__ int   si[256];
  const int p = blockIdx.x;
  float best = 3.0e38f; int bi = 0x7fffffff;
  for (int b = threadIdx.x; b < NBLK; b += 256) {
    float v = pval[(size_t)p * NBLK + b];
    int   i = pidx[(size_t)p * NBLK + b];
    if (v < best || (v == best && i < bi)) { best = v; bi = i; }
  }
  sv[threadIdx.x] = best; si[threadIdx.x] = bi;
  __syncthreads();
  for (int s = 128; s > 0; s >>= 1) {
    if (threadIdx.x < s) {
      float v = sv[threadIdx.x + s]; int i = si[threadIdx.x + s];
      if (v < sv[threadIdx.x] || (v == sv[threadIdx.x] && i < si[threadIdx.x])) {
        sv[threadIdx.x] = v; si[threadIdx.x] = i;
      }
    }
    __syncthreads();
  }
  if (threadIdx.x == 0) { minv[p] = sv[0]; mini[p] = si[0]; }
}

// ---------------------------------------------------------------------------
// 4) argmax of min_val -> s_star, s_idx, star library index
// ---------------------------------------------------------------------------
__global__ void k_argmax(const float* __restrict__ minv,
                         const int* __restrict__ mini,
                         float* __restrict__ sf, int* __restrict__ sI) {
  __shared__ float sv[256];
  __shared__ int   si[256];
  float best = -3.0e38f; int bp = 0x7fffffff;
  for (int p = threadIdx.x; p < PP; p += 256) {
    float v = minv[p];
    if (v > best || (v == best && p < bp)) { best = v; bp = p; }
  }
  sv[threadIdx.x] = best; si[threadIdx.x] = bp;
  __syncthreads();
  for (int s = 128; s > 0; s >>= 1) {
    if (threadIdx.x < s) {
      float v = sv[threadIdx.x + s]; int i = si[threadIdx.x + s];
      if (v > sv[threadIdx.x] || (v == sv[threadIdx.x] && i < si[threadIdx.x])) {
        sv[threadIdx.x] = v; si[threadIdx.x] = i;
      }
    }
    __syncthreads();
  }
  if (threadIdx.x == 0) {
    sf[0] = sv[0];            // s_star
    sI[0] = si[0];            // s_idx
    sI[1] = mini[si[0]];      // library index of m_star
  }
}

// ---------------------------------------------------------------------------
// 5) w_dist[n] = || lib[star] - lib[n] ||   (one wave per row, bandwidth bound)
// ---------------------------------------------------------------------------
__global__ void k_wdist(const float* __restrict__ lib,
                        const int* __restrict__ sI,
                        float* __restrict__ wdist) {
  const int n    = blockIdx.x * 8 + (threadIdx.x >> 5);
  const int lane = threadIdx.x & 31;
  if (n >= NLIB) return;
  const float* ms  = lib + (size_t)sI[1] * DD;
  const float* row = lib + (size_t)n * DD;
  float acc = 0.f;
  for (int j = lane; j < DD; j += 32) {
    float d = ms[j] - row[j];
    acc += d * d;
  }
  #pragma unroll
  for (int m = 1; m < 32; m <<= 1) acc += __shfl_xor(acc, m, 32);
  if (lane == 0) wdist[n] = sqrtf(acc);
}

// ---------------------------------------------------------------------------
// 6) top-3 smallest w_dist; keep indices 1 and 2 (skip m_star itself)
// ---------------------------------------------------------------------------
__global__ void k_top3(const float* __restrict__ wdist, int* __restrict__ sI) {
  __shared__ float sv[256 * 3];
  __shared__ int   si[256 * 3];
  float v0 = 3e38f, v1 = 3e38f, v2 = 3e38f;
  int   i0 = -1,    i1 = -1,    i2 = -1;
  for (int n = threadIdx.x; n < NLIB; n += 256) {
    float v = wdist[n];
    if (v < v0)      { v2=v1;i2=i1; v1=v0;i1=i0; v0=v;i0=n; }
    else if (v < v1) { v2=v1;i2=i1; v1=v;i1=n; }
    else if (v < v2) { v2=v;i2=n; }
  }
  sv[threadIdx.x*3+0]=v0; si[threadIdx.x*3+0]=i0;
  sv[threadIdx.x*3+1]=v1; si[threadIdx.x*3+1]=i1;
  sv[threadIdx.x*3+2]=v2; si[threadIdx.x*3+2]=i2;
  __syncthreads();
  if (threadIdx.x == 0) {
    float b0=3e38f,b1=3e38f,b2=3e38f; int j0=-1,j1=-1,j2=-1;
    for (int t = 0; t < 256*3; ++t) {
      float v = sv[t]; int i = si[t];
      if (i < 0) continue;
      if (v < b0 || (v == b0 && i < j0)) { b2=b1;j2=j1; b1=b0;j1=j0; b0=v;j0=i; }
      else if (v < b1 || (v == b1 && i < j1)) { b2=b1;j2=j1; b1=v;j1=i; }
      else if (v < b2 || (v == b2 && i < j2)) { b2=v;j2=i; }
    }
    sI[2] = j1;   // nn_idx[1]
    sI[3] = j2;   // nn_idx[2]
  }
}

// ---------------------------------------------------------------------------
// 7) scalar score s = w * s_star
// ---------------------------------------------------------------------------
__global__ void k_final_s(const float* __restrict__ patch,
                          const float* __restrict__ lib,
                          const float* __restrict__ sf,
                          const int* __restrict__ sI,
                          float* __restrict__ out) {
  __shared__ float r1[256], r2[256];
  const int sidx = sI[0], nn1 = sI[2], nn2 = sI[3];
  const float* mt = patch + (size_t)sidx * DD;
  const float* a  = lib + (size_t)nn1 * DD;
  const float* b  = lib + (size_t)nn2 * DD;
  float s1 = 0.f, s2 = 0.f;
  for (int j = threadIdx.x; j < DD; j += 256) {
    float d1 = mt[j] - a[j]; s1 += d1 * d1;
    float d2 = mt[j] - b[j]; s2 += d2 * d2;
  }
  r1[threadIdx.x] = s1; r2[threadIdx.x] = s2;
  __syncthreads();
  for (int s = 128; s > 0; s >>= 1) {
    if (threadIdx.x < s) {
      r1[threadIdx.x] += r1[threadIdx.x + s];
      r2[threadIdx.x] += r2[threadIdx.x + s];
    }
    __syncthreads();
  }
  if (threadIdx.x == 0) {
    float e1 = sqrtf(r1[0]), e2 = sqrtf(r2[0]);
    float Ds = sqrtf((float)DD);
    float ss = sf[0];
    float w  = 1.0f - expf(ss / Ds) / (expf(e1 / Ds) + expf(e2 / Ds));
    out[0] = w * ss;
  }
}

// ---------------------------------------------------------------------------
// 8) bilinear 28x28 -> 224x224 (half-pixel centers, edge clamp)
// ---------------------------------------------------------------------------
__global__ void k_bilinear(const float* __restrict__ minv,
                           float* __restrict__ map0) {
  const int idx = blockIdx.x * 256 + threadIdx.x;
  if (idx >= IMGN) return;
  const int oy = idx / IMGW, ox = idx % IMGW;
  const float sc = 28.0f / 224.0f;
  float sy = (oy + 0.5f) * sc - 0.5f;
  float sx = (ox + 0.5f) * sc - 0.5f;
  int y0 = (int)floorf(sy), x0 = (int)floorf(sx);
  float fy = sy - (float)y0, fx = sx - (float)x0;
  int y0c = min(max(y0, 0), 27),     y1c = min(max(y0 + 1, 0), 27);
  int x0c = min(max(x0, 0), 27),     x1c = min(max(x0 + 1, 0), 27);
  float v00 = minv[y0c*28 + x0c], v01 = minv[y0c*28 + x1c];
  float v10 = minv[y1c*28 + x0c], v11 = minv[y1c*28 + x1c];
  map0[idx] = (1.f-fy) * ((1.f-fx)*v00 + fx*v01)
            +      fy  * ((1.f-fx)*v10 + fx*v11);
}

// ---------------------------------------------------------------------------
// 9) separable Gaussian (sigma=4, radius=12, 25 taps), edge clamp
// ---------------------------------------------------------------------------
__global__ void k_blur(const float* __restrict__ in,
                       float* __restrict__ out, int horiz) {
  const int idx = blockIdx.x * 256 + threadIdx.x;
  if (idx >= IMGN) return;
  const int y = idx / IMGW, x = idx % IMGW;
  float num = 0.f, den = 0.f;
  for (int t = -12; t <= 12; ++t) {
    float ft = (float)t * 0.25f;              // t / sigma
    float w  = __expf(-0.5f * ft * ft);
    int yy = y, xx = x;
    if (horiz) xx = min(max(x + t, 0), IMGW - 1);
    else       yy = min(max(y + t, 0), IMGW - 1);
    num += w * in[yy * IMGW + xx];
    den += w;
  }
  out[idx] = num / den;
}

// ---------------------------------------------------------------------------
// launch
// ---------------------------------------------------------------------------
extern "C" void kernel_launch(void* const* d_in, const int* in_sizes, int n_in,
                              void* d_out, int out_size, void* d_ws, size_t ws_size,
                              hipStream_t stream) {
  const float* patch = (const float*)d_in[0];   // [784, 1536]
  const float* lib   = (const float*)d_in[1];   // [100000, 1536]
  float* out = (float*)d_out;                   // [1 + 224*224]
  char* ws = (char*)d_ws;

  size_t o = 0;
  auto take = [&](size_t bytes) -> char* {
    char* p = ws + o;
    o = (o + bytes + 255) & ~(size_t)255;
    return p;
  };
  __bf16* patch_bf = (__bf16*)take((size_t)PP * DD * 2);
  float*  x2       = (float*) take((size_t)PP * 4);
  float*  pval     = (float*) take((size_t)PP * NBLK * 4);
  int*    pidx     = (int*)   take((size_t)PP * NBLK * 4);
  float*  minv     = (float*) take((size_t)PP * 4);
  int*    mini     = (int*)   take((size_t)PP * 4);
  float*  wdist    = (float*) take((size_t)NLIB * 4);
  float*  map0     = (float*) take((size_t)IMGN * 4);
  float*  map1     = (float*) take((size_t)IMGN * 4);
  float*  sf       = (float*) take(64);
  int*    sI       = (int*)   take(64);
  (void)ws_size; (void)in_sizes; (void)n_in; (void)out_size;

  // 1) stage patch as bf16 + ||x||^2
  k_convert_patch<<<PP, 256, 0, stream>>>(patch, patch_bf, x2);

  // 2) WMMA distance GEMM + fused per-block min/argmin
  const size_t smem = 256 + (size_t)NCHUNK * LDS_ROW * 2;   // ~193.3 KB
  k_dist_min<<<NBLK, 256, smem, stream>>>(lib, patch_bf, x2, pval, pidx);

  // 3) reduce partials
  k_reduce_min<<<PP, 256, 0, stream>>>(pval, pidx, minv, mini);

  // 4) s_star / s_idx / star library row
  k_argmax<<<1, 256, 0, stream>>>(minv, mini, sf, sI);

  // 5) re-weight distances + 6) top-3 + 7) scalar score
  k_wdist<<<(NLIB + 7) / 8, 256, 0, stream>>>(lib, sI, wdist);
  k_top3<<<1, 256, 0, stream>>>(wdist, sI);
  k_final_s<<<1, 256, 0, stream>>>(patch, lib, sf, sI, out);

  // 8-9) anomaly map: bilinear upsample then separable Gaussian blur
  const int mblk = (IMGN + 255) / 256;
  k_bilinear<<<mblk, 256, 0, stream>>>(minv, map0);
  k_blur<<<mblk, 256, 0, stream>>>(map0, map1, 1);
  k_blur<<<mblk, 256, 0, stream>>>(map1, out + 1, 0);
}
